// TokenizedZeroConvPatchAttn_20486994002004
// MI455X (gfx1250) — matrix operand order repaired and checked by reading, not compiled, not run on hardware
//
#include <hip/hip_runtime.h>
#include <hip/hip_bf16.h>
#include <stdint.h>

// ---------------- problem constants ----------------
#define D_MODEL    768
#define B_IMG      64
#define KEEP16     118
#define KEEP32     20
#define TOK_STRIDE 139                  // 1 cls + 118 + 20 per image
#define T_TOK      (B_IMG * TOK_STRIDE) // 8896

typedef __bf16 v16bf  __attribute__((ext_vector_type(16)));
typedef __bf16 bf16x2 __attribute__((ext_vector_type(2)));
typedef float  v8f    __attribute__((ext_vector_type(8)));

// CDNA5 async global->LDS staging (ASYNCcnt) via inline asm:
//   global_load_async_to_lds_b128 vdst(lds byte offset), vaddr(64b), off
// LDS byte offset of a __shared__ object == low 32 bits of its generic pointer
// (ISA 10.2: LDS aperture truncates to addr[31:0]).
#define USE_ASYNC_LDS 1

#if defined(__has_builtin)
# if __has_builtin(__builtin_amdgcn_s_wait_asynccnt)
#  define WAIT_ASYNC(n) __builtin_amdgcn_s_wait_asynccnt(n)
# else
#  define WAIT_ASYNC(n) asm volatile("s_wait_asynccnt %0" ::"i"(n) : "memory")
# endif
#else
# define WAIT_ASYNC(n) asm volatile("s_wait_asynccnt %0" ::"i"(n) : "memory")
#endif

static __device__ __forceinline__ unsigned short f32_to_bf16(float f) {
  union { float f; unsigned u; } v; v.f = f;
  unsigned r = v.u + 0x7FFFu + ((v.u >> 16) & 1u);   // round-to-nearest-even
  return (unsigned short)(r >> 16);
}

// ---------------- small prep kernels ----------------
__global__ void cast_bf16_kernel(const float* __restrict__ src,
                                 unsigned short* __restrict__ dst, int n) {
  int i = blockIdx.x * blockDim.x + threadIdx.x;
  if (i < n) dst[i] = f32_to_bf16(src[i]);
}

// dst shape (C,R) = transpose of src shape (R,C), cast to bf16
__global__ void transpose_cast_kernel(const float* __restrict__ src,
                                      unsigned short* __restrict__ dst,
                                      int R, int C) {
  int i = blockIdx.x * blockDim.x + threadIdx.x;
  if (i < R * C) {
    int c = i / R, r = i % R;
    dst[i] = f32_to_bf16(src[(size_t)r * C + c]);
  }
}

// attn_w (e,d,h,w) -> W2T[e][(h*2+w)*768 + d]  (i.e. Bᵀ for the attn GEMM)
__global__ void permute_attn_kernel(const float* __restrict__ aw,
                                    unsigned short* __restrict__ dst) {
  int i = blockIdx.x * blockDim.x + threadIdx.x;
  if (i < 768 * 3072) {
    int e = i / 3072, r = i % 3072;
    int hw = r / 768, d = r % 768;
    int h = hw >> 1, w = hw & 1;
    dst[i] = f32_to_bf16(aw[(((size_t)e * 768 + d) * 2 + h) * 2 + w]);
  }
}

// per-image list of kept grid positions (prefix scan of the bool mask)
__global__ void build_gidx_kernel(const unsigned char* __restrict__ pm,
                                  int* __restrict__ gidx, int G, int Kkeep) {
  int b = blockIdx.x * blockDim.x + threadIdx.x;
  if (b >= B_IMG) return;
  int k = 0;
  for (int g = 0; g < G; ++g)
    if (pm[b * G + g] && k < Kkeep) gidx[b * Kkeep + k++] = g;
}

static __device__ __forceinline__ float cubic_w(float t) {
  t = fabsf(t);
  const float a = -0.5f;
  if (t <= 1.0f) return ((a + 2.0f) * t - (a + 3.0f)) * t * t + 1.0f;
  if (t < 2.0f)  return (((t - 5.0f) * t + 8.0f) * t - 4.0f) * a;
  return 0.0f;
}

// bicubic (antialiased, scale 2) resize of 14x14xD pos grid -> 7x7xD
__global__ void resize_pos32_kernel(const float* __restrict__ bpe,
                                    float* __restrict__ pos32) {
  int o = blockIdx.x;     // 0..48
  int d = threadIdx.x;    // 0..767
  int oy = o / 7, ox = o % 7;
  float wy[8], wx[8]; int iy[8], ix[8];
  float sy = 0.f, sx = 0.f;
#pragma unroll
  for (int t = 0; t < 8; ++t) {
    int i = 2 * oy - 3 + t;
    float w = cubic_w(((float)i - (2.0f * oy + 0.5f)) * 0.5f);
    if (i < 0 || i > 13) w = 0.f;
    iy[t] = min(max(i, 0), 13); wy[t] = w; sy += w;
    i = 2 * ox - 3 + t;
    w = cubic_w(((float)i - (2.0f * ox + 0.5f)) * 0.5f);
    if (i < 0 || i > 13) w = 0.f;
    ix[t] = min(max(i, 0), 13); wx[t] = w; sx += w;
  }
  const float* g = bpe + D_MODEL;  // skip cls row
  float s = 0.f;
#pragma unroll
  for (int ty = 0; ty < 8; ++ty) {
    float ra = 0.f;
#pragma unroll
    for (int tx = 0; tx < 8; ++tx)
      ra += wx[tx] * g[((size_t)iy[ty] * 14 + ix[tx]) * D_MODEL + d];
    s += wy[ty] * ra;
  }
  pos32[(size_t)o * D_MODEL + d] = s / (sy * sx);
}

// cls rows + attn_mask + cls_idx tails of d_out
__global__ void write_misc_kernel(const float* __restrict__ cls_token,
                                  const float* __restrict__ bpe,
                                  const unsigned char* __restrict__ amask,
                                  float* __restrict__ out) {
  int t = blockIdx.x * blockDim.x + threadIdx.x;
  if (t < B_IMG * D_MODEL) {
    int b = t / D_MODEL, n = t % D_MODEL;
    out[(size_t)b * TOK_STRIDE * D_MODEL + n] = cls_token[n] + bpe[n];
  }
  size_t maskOff = (size_t)T_TOK * D_MODEL;
  if (t < T_TOK) out[maskOff + t] = amask[t] ? 1.0f : 0.0f;
  if (t < B_IMG) out[maskOff + T_TOK + t] = (float)(t * TOK_STRIDE);
}

// ---------------- WMMA bf16 GEMM, N = 768, fused epilogue ----------------
#define BM 128
#define BN 128
#define KC 32
#define LDT_W   20      // 32 bf16 + 8 pad = 80 B = 20 dwords per LDS row
#define TILE_W  2560    // 128 rows * 20 dwords
#define BUF_W   5120    // A tile + B tile

__global__ __launch_bounds__(256)
void gemm_bf16_768(const unsigned short* __restrict__ A,   // (M,K) bf16 row-major
                   const unsigned short* __restrict__ BT,  // (N,K) bf16 row-major
                   int M, int K,
                   const float* __restrict__ bias,         // [768], always added
                   const float* __restrict__ posTab,       // rows of 768, or null
                   const int*   __restrict__ gidx,         // m -> pos row, or null
                   const float* __restrict__ addSrc,       // (M,768) f32, or null
                   float* __restrict__ outF,               // f32 out (or null)
                   unsigned short* __restrict__ outB,      // bf16 out (or null)
                   int rowsPerImg, int rowOffset)          // token scatter params
{
  __shared__ unsigned int smemU[2 * BUF_W];                // 40 KB
  const int tid  = threadIdx.x;
  const int lane = tid & 31;
  const int wave = tid >> 5;
  const int bn = blockIdx.x, bm = blockIdx.y;
  const int mo = (wave >> 1) * 32;      // wave's M offset in block tile
  const int no = (wave & 1) * 64;       // wave's N offset in block tile
  const int nK = K / KC;

  const unsigned short* Abase = A  + (size_t)bm * BM * K;
  const unsigned short* Bbase = BT + (size_t)bn * BN * K;

  // Each thread owns 4 disjoint 16B chunks per stage (2 tiles * 128 rows * 4).
  auto stage = [&](int kc, int buf) {
    const int kof = kc * KC;
#pragma unroll
    for (int i = 0; i < 4; ++i) {
      int idx = i * 256 + tid;
      int tsel = idx >> 9;            // 0 = A tile, 1 = B tile
      int r = (idx >> 2) & 127;
      int c = idx & 3;
      const unsigned short* src =
          (tsel ? Bbase : Abase) + (size_t)r * K + kof + c * 8;
      if (i == 0 && kof + KC < K)
        __builtin_prefetch(src + KC, 0, 1);   // global_prefetch_b8
      int w = buf * BUF_W + tsel * TILE_W + r * LDT_W + c * 4;
#ifdef USE_ASYNC_LDS
      unsigned lds_b = (unsigned)(uintptr_t)(&smemU[w]);   // LDS byte offset
      unsigned long long ga = (unsigned long long)(uintptr_t)src;
      asm volatile("global_load_async_to_lds_b128 %0, %1, off"
                   :: "v"(lds_b), "v"(ga) : "memory");
#else
      uint4 v = *(const uint4*)src;
      *(uint4*)&smemU[w] = v;
#endif
    }
  };

  v8f acc[2][4];
#pragma unroll
  for (int i = 0; i < 2; ++i)
#pragma unroll
    for (int j = 0; j < 4; ++j)
#pragma unroll
      for (int r = 0; r < 8; ++r) acc[i][j][r] = 0.0f;

  const int rsel = lane & 15;           // M (for A) / N (for B) within tile
  const int akb  = (lane >> 4) << 3;    // A: lanes 16-31 start at K=8
  const int bkb  = (lane >> 4) << 3;    // B: lanes 16-31 start at K=8

  stage(0, 0);

  for (int kc = 0; kc < nK; ++kc) {
    const int cur = kc & 1;
    if (kc + 1 < nK) {
      stage(kc + 1, cur ^ 1);   // 4 async ops into the other buffer
      WAIT_ASYNC(4);            // previous stage (buf cur) has landed
    } else {
      WAIT_ASYNC(0);
    }
    __syncthreads();            // buf cur visible to all waves

    const unsigned int* Asm = &smemU[cur * BUF_W];
    const unsigned int* Bsm = &smemU[cur * BUF_W + TILE_W];

    v16bf af[2], bfr[4];
#pragma unroll
    for (int i = 0; i < 2; ++i) {
      int row = mo + i * 16 + rsel;
#pragma unroll
      for (int j = 0; j < 8; ++j) {
        int k = (j < 4) ? (akb + 2 * j) : (akb + 16 + 2 * (j - 4));
        bf16x2 p = __builtin_bit_cast(bf16x2, Asm[row * LDT_W + (k >> 1)]);
        af[i][2 * j] = p[0]; af[i][2 * j + 1] = p[1];
      }
    }
#pragma unroll
    for (int j = 0; j < 4; ++j) {
      int row = no + j * 16 + rsel;
#pragma unroll
      for (int q = 0; q < 8; ++q) {
        bf16x2 p = __builtin_bit_cast(bf16x2, Bsm[row * LDT_W + ((bkb >> 1) + (q < 4 ? q : q + 4))]);
        bfr[j][2 * q] = p[0]; bfr[j][2 * q + 1] = p[1];
      }
    }
#pragma unroll
    for (int i = 0; i < 2; ++i)
#pragma unroll
      for (int j = 0; j < 4; ++j)
        acc[i][j] = __builtin_amdgcn_wmma_f32_16x16x32_bf16(
            false, af[i], false, bfr[j], (short)0, acc[i][j], false, false);

    __syncthreads();            // safe to overwrite buf cur next iteration
  }

  // fused epilogue: +bias, +pos gather, +residual, scatter/store
  const int mloc = (lane >> 4) << 3;
  const int nloc = lane & 15;
#pragma unroll
  for (int i = 0; i < 2; ++i) {
#pragma unroll
    for (int j = 0; j < 4; ++j) {
      int gn = bn * BN + no + j * 16 + nloc;
      float bv = bias[gn];
#pragma unroll
      for (int r = 0; r < 8; ++r) {
        int gm = bm * BM + mo + i * 16 + mloc + r;
        float v = acc[i][j][r] + bv;
        if (gidx)   v += posTab[(size_t)gidx[gm] * D_MODEL + gn];
        if (addSrc) v += addSrc[(size_t)gm * D_MODEL + gn];
        if (outB) {
          outB[(size_t)gm * D_MODEL + gn] = f32_to_bf16(v);
        } else {
          size_t row = rowsPerImg
              ? (size_t)(gm / rowsPerImg) * TOK_STRIDE + rowOffset + (gm % rowsPerImg)
              : (size_t)gm;
          outF[row * D_MODEL + gn] = v;
        }
      }
    }
  }
}

// ---------------- launcher ----------------
extern "C" void kernel_launch(void* const* d_in, const int* in_sizes, int n_in,
                              void* d_out, int out_size, void* d_ws, size_t ws_size,
                              hipStream_t stream) {
  (void)in_sizes; (void)n_in; (void)out_size; (void)ws_size;
  const float* bpe     = (const float*)d_in[1];
  const float* rp16    = (const float*)d_in[2];
  const float* rp32    = (const float*)d_in[3];
  const float* fpatch  = (const float*)d_in[4];
  const float* patch_w = (const float*)d_in[5];
  const float* patch_b = (const float*)d_in[6];
  const float* cls_tok = (const float*)d_in[7];
  const float* attn_w  = (const float*)d_in[8];
  const float* attn_b  = (const float*)d_in[9];
  const float* zero_w  = (const float*)d_in[10];
  const float* zero_b  = (const float*)d_in[11];
  const unsigned char* pm16  = (const unsigned char*)d_in[13];
  const unsigned char* pm32  = (const unsigned char*)d_in[14];
  const unsigned char* amask = (const unsigned char*)d_in[15];
  float* outF = (float*)d_out;

  const int N16 = B_IMG * KEEP16;   // 7552
  const int N32 = B_IMG * KEEP32;   // 1280
  const int NF  = 4 * N32;          // 5120

  char* ws = (char*)d_ws;
  size_t off = 0;
  auto alloc = [&](size_t bytes) -> void* {
    off = (off + 255) & ~(size_t)255;
    void* p = ws + off; off += bytes; return p;
  };
  unsigned short* A16 = (unsigned short*)alloc((size_t)N16 * 768 * 2);
  unsigned short* A32 = (unsigned short*)alloc((size_t)N32 * 768 * 2);
  unsigned short* AF  = (unsigned short*)alloc((size_t)NF  * 768 * 2);
  unsigned short* WpT = (unsigned short*)alloc((size_t)768 * 768 * 2);
  unsigned short* W2T = (unsigned short*)alloc((size_t)768 * 3072 * 2);
  unsigned short* WzT = (unsigned short*)alloc((size_t)768 * 768 * 2);
  unsigned short* FPb = (unsigned short*)alloc((size_t)NF  * 768 * 2);
  unsigned short* ATb = (unsigned short*)alloc((size_t)N32 * 768 * 2);
  float* ES  = (float*)alloc((size_t)N32 * 768 * 4);
  float* P32 = (float*)alloc((size_t)49 * 768 * 4);
  int* g16 = (int*)alloc((size_t)N16 * 4);
  int* g32 = (int*)alloc((size_t)N32 * 4);

  auto cdiv = [](long a, long b) { return (int)((a + b - 1) / b); };
  int n;
  n = N16 * 768; cast_bf16_kernel<<<cdiv(n,256),256,0,stream>>>(rp16, A16, n);
  n = N32 * 768; cast_bf16_kernel<<<cdiv(n,256),256,0,stream>>>(rp32, A32, n);
  n = NF  * 768; cast_bf16_kernel<<<cdiv(n,256),256,0,stream>>>(fpatch, AF, n);
  n = 768 * 768; transpose_cast_kernel<<<cdiv(n,256),256,0,stream>>>(patch_w, WpT, 768, 768);
  n = 768 * 3072; permute_attn_kernel<<<cdiv(n,256),256,0,stream>>>(attn_w, W2T);
  n = 768 * 768; cast_bf16_kernel<<<cdiv(n,256),256,0,stream>>>(zero_w, WzT, n);

  resize_pos32_kernel<<<49, 768, 0, stream>>>(bpe, P32);
  build_gidx_kernel<<<1, 64, 0, stream>>>(pm16, g16, 196, KEEP16);
  build_gidx_kernel<<<1, 64, 0, stream>>>(pm32, g32, 49, KEEP32);
  write_misc_kernel<<<cdiv(B_IMG * D_MODEL, 256), 256, 0, stream>>>(cls_tok, bpe, amask, outF);

  dim3 blk(256);
  // embed16 -> scatter to token rows (pos gather from base grid, row 1+g)
  gemm_bf16_768<<<dim3(6, N16 / BM), blk, 0, stream>>>(
      A16, WpT, N16, 768, patch_b, bpe + D_MODEL, g16, nullptr,
      outF, nullptr, KEEP16, 1);
  // embed_scale (f32 to ws, consumed as residual later)
  gemm_bf16_768<<<dim3(6, N32 / BM), blk, 0, stream>>>(
      A32, WpT, N32, 768, patch_b, nullptr, nullptr, nullptr,
      ES, nullptr, 0, 0);
  // fp = patch_embed(full patches), bf16 to ws (feeds attn GEMM)
  gemm_bf16_768<<<dim3(6, NF / BM), blk, 0, stream>>>(
      AF, WpT, NF, 768, patch_b, nullptr, nullptr, nullptr,
      nullptr, FPb, 0, 0);
  // attn_scale = fp_flat @ W2 + attn_b   (K = 3072), bf16 to ws
  gemm_bf16_768<<<dim3(6, N32 / BM), blk, 0, stream>>>(
      FPb, W2T, N32, 3072, attn_b, nullptr, nullptr, nullptr,
      nullptr, ATb, 0, 0);
  // final: attn_scale @ zero_wᵀ + zero_b + embed_scale + pos32 -> token rows
  gemm_bf16_768<<<dim3(6, N32 / BM), blk, 0, stream>>>(
      ATb, WzT, N32, 768, zero_b, P32, g32, ES,
      outF, nullptr, KEEP32, 119);
}
